// Decoder_80736795230986
// MI455X (gfx1250) — matrix operand order repaired and checked by reading, not compiled, not run on hardware
//
#include <hip/hip_runtime.h>

// ---------------------------------------------------------------------------
// SNN (2-layer LIF, T=16) for MI455X / gfx1250, wave32 + WMMA bf16.
//   layer1: h1 = v1 + (x@W1^T - v1)/2 ; s1 = h1>=0.5 ; v1 = h1*(1-s1)
//   layer2: h2 = v2 + (s1@Wout^T - v2)/2 ; s2 ; votes = sum_t avgpool10(s2)/16
// d_out = [out_spikes (4096*11 f32)] ++ [de_spk = s1 (4096*16*1024 f32)]
// ---------------------------------------------------------------------------

typedef __attribute__((ext_vector_type(16))) __bf16   v16bf;
typedef __attribute__((ext_vector_type(8)))  float    v8f;
typedef __attribute__((ext_vector_type(8)))  unsigned v8u;

#define SNN_B    4096
#define SNN_T    16
#define SNN_D    512
#define SNN_H    1024
#define SNN_NOUT 110
#define SNN_NCLS 11

// fp32 -> bf16 (round to nearest even), returned as raw u16
static __device__ __forceinline__ unsigned short f2bf_u(float f) {
  unsigned u = __builtin_bit_cast(unsigned, f);
  u += 0x7FFFu + ((u >> 16) & 1u);
  return (unsigned short)(u >> 16);
}
static __device__ __forceinline__ unsigned pack_bf2(float a, float b) {
  return (unsigned)f2bf_u(a) | ((unsigned)f2bf_u(b) << 16);
}

// ---------------------------------------------------------------------------
// Layer 1: per WG a 32(B-rows) x 32(H-cols) tile, all 16 timesteps.
// W1 slice staged to LDS (bf16) once; x tile staged per step.
// Each of 4 waves holds one 16x16 C-tile; v1 lives in its 8 accum VGPRs.
// ---------------------------------------------------------------------------
__global__ __launch_bounds__(128) void snn_layer1(
    const float* __restrict__ x,      // [B,T,D]
    const float* __restrict__ W1,     // [H,D]
    float* __restrict__ s1out)        // [B,T,H] (= de_spk region of d_out)
{
  // rows padded to 260 uints (=520 bf16) -> row stride 4 banks apart
  __shared__ unsigned Wsh[32][260];   // W1[n0..n0+31][0..511] as bf16
  __shared__ unsigned Ash[32][260];   // x_t[b0..b0+31][0..511] as bf16

  const int b0   = blockIdx.x * 32;
  const int n0   = blockIdx.y * 32;
  const int tid  = threadIdx.x;
  const int lane = tid & 31;
  const int wave = tid >> 5;          // 0..3
  const int m0   = (wave & 1) * 16;   // B sub-tile
  const int nn0  = (wave >> 1) * 16;  // H sub-tile
  const int lidx = lane & 15;
  const int half = lane >> 4;

  // stage W1 slice once (coalesced float4 reads, packed bf16 dword LDS writes)
  for (int i = tid; i < 32 * 128; i += 128) {
    const int row = i >> 7;
    const int c4  = i & 127;
    const float4 w4 = ((const float4*)(W1 + (size_t)(n0 + row) * SNN_D))[c4];
    Wsh[row][c4 * 2 + 0] = pack_bf2(w4.x, w4.y);
    Wsh[row][c4 * 2 + 1] = pack_bf2(w4.z, w4.w);
  }

  v8f v1 = {0.f, 0.f, 0.f, 0.f, 0.f, 0.f, 0.f, 0.f};

  for (int t = 0; t < SNN_T; ++t) {
    __syncthreads();  // previous step's fragment reads done before overwrite
    for (int i = tid; i < 32 * 128; i += 128) {
      const int row = i >> 7;
      const int c4  = i & 127;
      const float4 a4 =
          ((const float4*)(x + ((size_t)(b0 + row) * SNN_T + t) * SNN_D))[c4];
      Ash[row][c4 * 2 + 0] = pack_bf2(a4.x, a4.y);
      Ash[row][c4 * 2 + 1] = pack_bf2(a4.z, a4.w);
    }
    if (t + 1 < SNN_T) {  // hint next step's x tile toward the caches
      __builtin_prefetch(
          x + ((size_t)(b0 + (tid >> 2)) * SNN_T + (t + 1)) * SNN_D +
              (size_t)(tid & 3) * 128, 0, 1);
    }
    __syncthreads();

    v8f acc = {0.f, 0.f, 0.f, 0.f, 0.f, 0.f, 0.f, 0.f};
#pragma unroll
    for (int kc = 0; kc < SNN_D; kc += 32) {
      v8u ai, bi;
#pragma unroll
      for (int q = 0; q < 8; ++q) {
        // A (16x32 bf16): lane m = lidx; K pairs per ISA 7.12.2
        const int ka = kc + ((q & 3) * 2 + half * 8 + (q >> 2) * 16);
        ai[q] = Ash[m0 + lidx][ka >> 1];
        // B (32x16 bf16): lane n = lidx; K pairs same order as A
        const int kb = kc + ((q & 3) * 2 + half * 8 + (q >> 2) * 16);
        bi[q] = Wsh[nn0 + lidx][kb >> 1];
      }
      acc = __builtin_amdgcn_wmma_f32_16x16x32_bf16(
          false, __builtin_bit_cast(v16bf, ai),
          false, __builtin_bit_cast(v16bf, bi),
          (short)0, acc, false, false);
    }

    // LIF update in fp32; spike straight out of the C layout
#pragma unroll
    for (int r = 0; r < 8; ++r) {
      const float h = v1[r] + (acc[r] - v1[r]) * 0.5f;   // 1/TAU = 0.5
      const float s = (h >= 0.5f) ? 1.0f : 0.0f;         // VTH = 0.5
      v1[r] = h * (1.0f - s);                            // hard reset
      const int gb = b0 + m0 + r + 8 * half;
      const int gh = n0 + nn0 + lidx;
      s1out[((size_t)gb * SNN_T + t) * SNN_H + gh] = s;
    }
  }
}

// ---------------------------------------------------------------------------
// Layer 2: per WG a 16-row B-strip for all t. 8 waves: waves 0..6 each own one
// 16-col tile of the (110 -> 112 padded) outputs; wave 7 only helps staging.
// s1 strip staged to LDS per step; Wout (450 KB) read per-lane from L2.
// ---------------------------------------------------------------------------
__global__ __launch_bounds__(256) void snn_layer2(
    const float* __restrict__ s1,     // [B,T,H] (de_spk region)
    const float* __restrict__ Wout,   // [110,H]
    float* __restrict__ votebuf)      // [B,110] spike counts over t
{
  __shared__ unsigned Ash[16][516];   // s1 strip as bf16 (16 x 1024)

  const int b0     = blockIdx.x * 16;
  const int tid    = threadIdx.x;
  const int lane   = tid & 31;
  const int wave   = tid >> 5;        // 0..7
  const int lidx   = lane & 15;
  const int half   = lane >> 4;
  const int n      = wave * 16 + lidx;                       // 0..127
  const int nclamp = (n < SNN_NOUT) ? n : (SNN_NOUT - 1);    // avoid OOB reads

  v8f v2    = {0.f, 0.f, 0.f, 0.f, 0.f, 0.f, 0.f, 0.f};
  v8f votes = {0.f, 0.f, 0.f, 0.f, 0.f, 0.f, 0.f, 0.f};

  for (int t = 0; t < SNN_T; ++t) {
    __syncthreads();
    for (int i = tid; i < 16 * 256; i += 256) {
      const int row = i >> 8;
      const int c4  = i & 255;
      const float4 a4 =
          ((const float4*)(s1 + ((size_t)(b0 + row) * SNN_T + t) * SNN_H))[c4];
      Ash[row][c4 * 2 + 0] = pack_bf2(a4.x, a4.y);
      Ash[row][c4 * 2 + 1] = pack_bf2(a4.z, a4.w);
    }
    __syncthreads();

    if (wave < 7) {  // wave-uniform: EXEC all-ones inside (WMMA requirement)
      v8f acc = {0.f, 0.f, 0.f, 0.f, 0.f, 0.f, 0.f, 0.f};
#pragma unroll 4
      for (int kc = 0; kc < SNN_H; kc += 32) {
        v8u ai, bi;
#pragma unroll
        for (int q = 0; q < 8; ++q) {
          const int ka = kc + ((q & 3) * 2 + half * 8 + (q >> 2) * 16);
          ai[q] = Ash[lidx][ka >> 1];
          const int kb = kc + ((q & 3) * 2 + half * 8 + (q >> 2) * 16);
          const float2 w2 =
              *(const float2*)(Wout + (size_t)nclamp * SNN_H + kb);
          bi[q] = pack_bf2(w2.x, w2.y);
        }
        acc = __builtin_amdgcn_wmma_f32_16x16x32_bf16(
            false, __builtin_bit_cast(v16bf, ai),
            false, __builtin_bit_cast(v16bf, bi),
            (short)0, acc, false, false);
      }
#pragma unroll
      for (int r = 0; r < 8; ++r) {
        const float h = v2[r] + (acc[r] - v2[r]) * 0.5f;
        const float s = (h >= 0.5f) ? 1.0f : 0.0f;
        v2[r] = h * (1.0f - s);
        votes[r] += s;  // accumulate spike count over time
      }
    }
  }

  if (wave < 7 && n < SNN_NOUT) {
#pragma unroll
    for (int r = 0; r < 8; ++r) {
      votebuf[(size_t)(b0 + r + 8 * half) * SNN_NOUT + n] = votes[r];
    }
  }
}

// ---------------------------------------------------------------------------
// Vote pooling: out_spikes[b][c] = sum_{j<10} votes[b][10c+j] / (10*16)
// ---------------------------------------------------------------------------
__global__ __launch_bounds__(256) void snn_vote(
    const float* __restrict__ votebuf, float* __restrict__ out) {
  const int i = blockIdx.x * blockDim.x + threadIdx.x;
  if (i >= SNN_B * SNN_NCLS) return;
  const int b = i / SNN_NCLS;
  const int c = i % SNN_NCLS;
  float s = 0.f;
#pragma unroll
  for (int j = 0; j < 10; ++j)
    s += votebuf[(size_t)b * SNN_NOUT + c * 10 + j];
  out[i] = s * (1.0f / 160.0f);
}

// ---------------------------------------------------------------------------
extern "C" void kernel_launch(void* const* d_in, const int* in_sizes, int n_in,
                              void* d_out, int out_size, void* d_ws,
                              size_t ws_size, hipStream_t stream) {
  (void)in_sizes; (void)n_in; (void)out_size; (void)ws_size;
  const float* x    = (const float*)d_in[0];   // [4096,16,512]
  const float* W1   = (const float*)d_in[1];   // [1024,512]
  const float* Wout = (const float*)d_in[2];   // [110,1024]

  float* out_spikes = (float*)d_out;                       // [4096,11]
  float* de_spk     = (float*)d_out + SNN_B * SNN_NCLS;    // [4096,16,1024]
  float* votebuf    = (float*)d_ws;                        // [4096,110]

  dim3 g1(SNN_B / 32, SNN_H / 32);
  snn_layer1<<<g1, 128, 0, stream>>>(x, W1, de_spk);

  snn_layer2<<<SNN_B / 16, 256, 0, stream>>>(de_spk, Wout, votebuf);

  const int nvote = SNN_B * SNN_NCLS;
  snn_vote<<<(nvote + 255) / 256, 256, 0, stream>>>(votebuf, out_spikes);
}
